// VectorQuantizer_38439957299885
// MI455X (gfx1250) — compile-verified, matrix operand order checked by reading, not run on hardware
//
#include <hip/hip_runtime.h>
#include <cstdint>
#include <cstddef>

typedef __attribute__((ext_vector_type(16))) __bf16          v16bf;
typedef __attribute__((ext_vector_type(8)))  float           v8f;
typedef __attribute__((ext_vector_type(8)))  unsigned short  us8;

#define DDIM    64
#define THREADS 256
#define WAVES   8      // 256 threads / wave32

__device__ __forceinline__ unsigned short f2bf_bits(float x) {
    unsigned u = __float_as_uint(x);
    u += 0x7FFFu + ((u >> 16) & 1u);          // round-to-nearest-even
    return (unsigned short)(u >> 16);
}
__device__ __forceinline__ float bf_bits2f(unsigned short h) {
    return __uint_as_float(((unsigned)h) << 16);
}

union UFrag { v16bf v; unsigned short s[16]; us8 u2[2]; };

// One wave owns a 16-row tile of z_e. Whole codebook lives in LDS as split
// bf16 (hi/lo) + f32 norms. dot(z,c) computed as 3x v_wmma_f32_16x16x32_bf16
// (hi*hi + hi*lo + lo*hi) per K=32 chunk -> ~f32 accuracy on the argmin.
__global__ __launch_bounds__(THREADS) void vq_main(
    const float* __restrict__ z_e, const float* __restrict__ codebook,
    float* __restrict__ out_zqst, float* __restrict__ out_zq,
    float* __restrict__ out_idx,  float* __restrict__ loss_partials,
    int N, int K, int numTiles)
{
    extern __shared__ char smem[];
    unsigned short* ch    = (unsigned short*)smem;    // K*64 bf16 hi
    unsigned short* cl    = ch + K * DDIM;            // K*64 bf16 lo
    float*          cnorm = (float*)(cl + K * DDIM);  // K norms
    float*          red   = cnorm + K;                // THREADS loss scratch

    const int tid = threadIdx.x;

    // ---- stage codebook into LDS as split bf16 + compute ||c||^2 in f32 ----
    for (int e = tid; e < K * DDIM; e += THREADS) {
        float v = codebook[e];
        unsigned short h = f2bf_bits(v);
        ch[e] = h;
        cl[e] = f2bf_bits(v - bf_bits2f(h));
    }
    for (int r = tid; r < K; r += THREADS) {
        const float* c = codebook + r * DDIM;
        float s = 0.f;
        #pragma unroll
        for (int j = 0; j < DDIM; ++j) s = fmaf(c[j], c[j], s);
        cnorm[r] = s;
    }
    __syncthreads();

    const int lane = tid & 31;
    const int wave = tid >> 5;
    const int m    = lane & 15;
    const int half = lane >> 4;   // wmma 16-bit operand K-offset group
    float lsum = 0.f;

    for (int tile = blockIdx.x * WAVES + wave; tile < numTiles;
         tile += gridDim.x * WAVES) {
        const int rowBase = tile * 16;

        // ---- A tile: 16 rows x 64 dims f32 -> split bf16 fragments ----
        int arow = rowBase + m; if (arow > N - 1) arow = N - 1;
        const float* zr = z_e + (size_t)arow * DDIM;
        UFrag ahi[2], alo[2];
        #pragma unroll
        for (int c2 = 0; c2 < 2; ++c2) {
            const int k0 = c2 * 32 + half * 8;
            const float4 p0 = *(const float4*)(zr + k0);
            const float4 p1 = *(const float4*)(zr + k0 + 4);
            const float4 p2 = *(const float4*)(zr + k0 + 16);
            const float4 p3 = *(const float4*)(zr + k0 + 20);
            float av[16] = { p0.x,p0.y,p0.z,p0.w, p1.x,p1.y,p1.z,p1.w,
                             p2.x,p2.y,p2.z,p2.w, p3.x,p3.y,p3.z,p3.w };
            #pragma unroll
            for (int j = 0; j < 16; ++j) {
                unsigned short h = f2bf_bits(av[j]);
                ahi[c2].s[j] = h;
                alo[c2].s[j] = f2bf_bits(av[j] - bf_bits2f(h));
            }
        }

        float bv[8]; int bi[8];
        #pragma unroll
        for (int j = 0; j < 8; ++j) { bv[j] = 3.402823466e38f; bi[j] = 0; }

        // ---- sweep all K codes in 16-wide column tiles ----
        const int numCT = K >> 4;
        for (int ct = 0; ct < numCT; ++ct) {
            const int crow = ct * 16 + m;     // this lane's codebook row (N dim)
            v8f acc = {0.f,0.f,0.f,0.f,0.f,0.f,0.f,0.f};
            #pragma unroll
            for (int c2 = 0; c2 < 2; ++c2) {
                const int off = crow * DDIM + c2 * 32 + half * 8;
                UFrag bh, bl;
                bh.u2[0] = *(const us8*)&ch[off];
                bh.u2[1] = *(const us8*)&ch[off + 16];
                bl.u2[0] = *(const us8*)&cl[off];
                bl.u2[1] = *(const us8*)&cl[off + 16];
                acc = __builtin_amdgcn_wmma_f32_16x16x32_bf16(
                          false, ahi[c2].v, false, bh.v, (short)0, acc, false, false);
                acc = __builtin_amdgcn_wmma_f32_16x16x32_bf16(
                          false, ahi[c2].v, false, bl.v, (short)0, acc, false, false);
                acc = __builtin_amdgcn_wmma_f32_16x16x32_bf16(
                          false, alo[c2].v, false, bh.v, (short)0, acc, false, false);
            }
            const float cn = cnorm[crow];
            #pragma unroll
            for (int j = 0; j < 8; ++j) {     // dist = ||c||^2 - 2*dot
                float d = fmaf(-2.f, acc[j], cn);
                if (d < bv[j]) { bv[j] = d; bi[j] = crow; }
            }
        }

        // ---- argmin butterfly across the 16 lanes of each half (wave32) ----
        #pragma unroll
        for (int mask = 1; mask < 16; mask <<= 1) {
            #pragma unroll
            for (int j = 0; j < 8; ++j) {
                float ov = __shfl_xor(bv[j], mask, 32);
                int   oi = __shfl_xor(bi[j], mask, 32);
                if (ov < bv[j] || (ov == bv[j] && oi < bi[j])) { bv[j] = ov; bi[j] = oi; }
            }
        }

        // ---- outputs: 2 lanes per row, 32 columns each ----
        const int rl = m >> 1;
        int myIdx = 0;
        #pragma unroll
        for (int j = 0; j < 8; ++j) if (j == rl) myIdx = bi[j];
        const int r = rowBase + half * 8 + rl;
        if (r < N) {
            const int c0 = (lane & 1) * 32;
            if ((lane & 1) == 0) out_idx[r] = (float)myIdx;
            const float* cb = codebook + (size_t)myIdx * DDIM + c0;
            const float* ze = z_e      + (size_t)r     * DDIM + c0;
            float* o0 = out_zqst + (size_t)r * DDIM + c0;
            float* o1 = out_zq   + (size_t)r * DDIM + c0;
            #pragma unroll
            for (int t = 0; t < 8; ++t) {
                float4 cq = *(const float4*)(cb + t * 4);
                float4 zv = *(const float4*)(ze + t * 4);
                float dx = cq.x - zv.x, dy = cq.y - zv.y;
                float dz = cq.z - zv.z, dw = cq.w - zv.w;
                float4 st = { zv.x + dx, zv.y + dy, zv.z + dz, zv.w + dw };
                *(float4*)(o1 + t * 4) = cq;   // z_q
                *(float4*)(o0 + t * 4) = st;   // z_q_st (== z_e + (z_q - z_e))
                lsum += dx*dx + dy*dy + dz*dz + dw*dw;
            }
        }
    }

    // ---- deterministic per-block loss partial ----
    red[tid] = lsum;
    __syncthreads();
    for (int s = THREADS / 2; s > 0; s >>= 1) {
        if (tid < s) red[tid] += red[tid + s];
        __syncthreads();
    }
    if (tid == 0) loss_partials[blockIdx.x] = red[0];
}

__global__ void vq_loss_reduce(const float* __restrict__ partials, int n,
                               float* __restrict__ out_loss, double scale)
{
    __shared__ double sd[256];
    double s = 0.0;
    for (int i = threadIdx.x; i < n; i += 256) s += (double)partials[i];
    sd[threadIdx.x] = s;
    __syncthreads();
    for (int st = 128; st > 0; st >>= 1) {
        if ((int)threadIdx.x < st) sd[threadIdx.x] += sd[threadIdx.x + st];
        __syncthreads();
    }
    if (threadIdx.x == 0) out_loss[0] = (float)(sd[0] * scale);
}

extern "C" void kernel_launch(void* const* d_in, const int* in_sizes, int n_in,
                              void* d_out, int out_size, void* d_ws, size_t ws_size,
                              hipStream_t stream)
{
    const float* z_e      = (const float*)d_in[0];
    const float* codebook = (const float*)d_in[1];
    const int D = DDIM;
    const int N = in_sizes[0] / D;   // 500000
    const int K = in_sizes[1] / D;   // 512

    float* out      = (float*)d_out;
    float* out_zqst = out;
    float* out_zq   = out + (size_t)N * D;
    float* out_idx  = out_zq + (size_t)N * D;
    float* out_loss = out_idx + N;
    float* partials = (float*)d_ws;

    const int numTiles = (N + 15) / 16;
    int blocks = (numTiles + WAVES - 1) / WAVES;
    if (blocks > 1024) blocks = 1024;
    if (ws_size && blocks > (int)(ws_size / sizeof(float)))
        blocks = (int)(ws_size / sizeof(float));
    if (blocks < 1) blocks = 1;

    // LDS: bf16 hi + lo codebook (2*K*64*2B) + f32 norms + loss scratch
    size_t shmem = (size_t)K * D * 2 * sizeof(unsigned short)
                 + (size_t)K * sizeof(float)
                 + THREADS * sizeof(float);   // = 134,144 B for K=512 (<320KB WGP LDS)

    vq_main<<<blocks, THREADS, shmem, stream>>>(
        z_e, codebook, out_zqst, out_zq, out_idx, partials, N, K, numTiles);

    const double scale = 1.25 / ((double)N * (double)D);
    vq_loss_reduce<<<1, 256, 0, stream>>>(partials, blocks, out_loss, scale);
}
